// MultiHeadAttentionWithRoPE_66391604461996
// MI455X (gfx1250) — matrix operand order repaired and checked by reading, not compiled, not run on hardware
//
#include <hip/hip_runtime.h>
#include <hip/hip_bf16.h>

// ---------- types ----------
typedef __attribute__((ext_vector_type(16))) __bf16        v16bf;
typedef __attribute__((ext_vector_type(8)))  float         v8f;
typedef __attribute__((ext_vector_type(4))) unsigned int   u32x4;
typedef __attribute__((ext_vector_type(2))) unsigned int   u32x2;
typedef __attribute__((ext_vector_type(4)))  float         f32x4;

union Frag {           // one WMMA A/B operand: 16 bf16 per lane = 2x 128-bit
    v16bf v;
    u32x4 q[2];
};
union Pack4 {          // 64-bit <-> 4 bf16
    u32x2 u;
    __bf16 h[4];
};
union Pack8 {          // 128-bit <-> 8 bf16
    u32x4 u;
    __bf16 h[8];
};

// Problem constants (B=2, S=2048, D=1024, H=16, d_k=64)
#define PB   2
#define PS   2048
#define PD   1024
#define PH   16
#define PDK  64
#define PM   (PB * PS)          // 4096 rows

// softmax runs in exp2 domain; Q is pre-scaled by (1/sqrt(64)) * log2(e)
#define QSCALE (0.125f * 1.4426950408889634f)

static __device__ __forceinline__ v8f wmma_bf16(v16bf a, v16bf b, v8f c) {
    return __builtin_amdgcn_wmma_f32_16x16x32_bf16(
        /*neg_a=*/false, a, /*neg_b=*/false, b,
        /*c_mod=*/(short)0, c, /*reuse_a=*/false, /*reuse_b=*/false);
}

// xor-lane exchange via DS_SWIZZLE_B32 (group-of-32 mode: and=0x1f, or=0, xor=XM)
template <int XM>
static __device__ __forceinline__ float swz_xor(float v) {
    int i = __builtin_amdgcn_ds_swizzle(__builtin_bit_cast(int, v),
                                        0x1f | (XM << 10));
    return __builtin_bit_cast(float, i);
}
static __device__ __forceinline__ float redmax16(float v) {
    v = fmaxf(v, swz_xor<8>(v));
    v = fmaxf(v, swz_xor<4>(v));
    v = fmaxf(v, swz_xor<2>(v));
    v = fmaxf(v, swz_xor<1>(v));
    return v;
}
static __device__ __forceinline__ float redsum16(float v) {
    v += swz_xor<8>(v);
    v += swz_xor<4>(v);
    v += swz_xor<2>(v);
    v += swz_xor<1>(v);
    return v;
}

// ---------- kernel 0: fp32 -> bf16 convert (vectorized x4) ----------
__global__ void cvt_f32_to_bf16x4(const float* __restrict__ in,
                                  __bf16* __restrict__ out, int n4) {
    int i = blockIdx.x * blockDim.x + threadIdx.x;
    if (i < n4) {
        f32x4 f = *(const f32x4*)(in + 4 * (size_t)i);
        Pack4 p;
        p.h[0] = (__bf16)f.x; p.h[1] = (__bf16)f.y;
        p.h[2] = (__bf16)f.z; p.h[3] = (__bf16)f.w;
        *(u32x2*)(out + 4 * (size_t)i) = p.u;
    }
}

// ---------- kernel 1/4: tiled WMMA GEMM  Y[M,N] = A[M,K] * W[N,K]^T ----------
// block = 256 threads (8 waves), tile 128(M) x 128(N), K-step 32.
// wave (wm in 0..3, wn in 0..1) computes a 32x64 patch = 2x4 WMMA tiles.
#define GBM 128
#define GBN 128
template <bool F32OUT>
__global__ __launch_bounds__(256, 1)
void gemm_bf16_wmma(const __bf16* __restrict__ A,
                    const __bf16* __restrict__ W0,
                    const __bf16* __restrict__ W1,
                    const __bf16* __restrict__ W2,
                    void* __restrict__ O0,
                    void* __restrict__ O1,
                    void* __restrict__ O2,
                    int M, int N, int K) {
    __shared__ __bf16 As[GBM * 32];   // 8 KB
    __shared__ __bf16 Bs[GBN * 32];   // 8 KB

    const __bf16* W = (blockIdx.z == 0) ? W0 : (blockIdx.z == 1 ? W1 : W2);
    void*         O = (blockIdx.z == 0) ? O0 : (blockIdx.z == 1 ? O1 : O2);

    const int mb  = blockIdx.y * GBM;
    const int nb  = blockIdx.x * GBN;
    const int tid = threadIdx.x;
    const int lane = tid & 31;
    const int l15  = lane & 15;
    const int hi   = lane >> 4;
    const int w    = tid >> 5;
    const int wm   = w & 3;   // 0..3  (M direction, 32 rows each)
    const int wn   = w >> 2;  // 0..1  (N direction, 64 cols each)

    v8f c[2][4] = {};

    for (int kb = 0; kb < K; kb += 32) {
        __syncthreads();
        // stage A/B tiles 128x32 with 128-bit loads (512 xb128 each)
        for (int i = tid; i < GBM * 4; i += 256) {
            int r = i >> 2, qw = i & 3;
            ((u32x4*)As)[r * 4 + qw] =
                *(const u32x4*)(A + (size_t)(mb + r) * K + kb + qw * 8);
        }
        for (int i = tid; i < GBN * 4; i += 256) {
            int r = i >> 2, qw = i & 3;
            ((u32x4*)Bs)[r * 4 + qw] =
                *(const u32x4*)(W + (size_t)(nb + r) * K + kb + qw * 8);
        }
        // prefetch next A tile into cache hierarchy (global_prefetch_b8)
        if (kb + 32 < K)
            __builtin_prefetch(A + (size_t)(mb + (tid >> 1)) * K + kb + 32, 0, 1);
        __syncthreads();

        Frag af[2], bf2[4];
#pragma unroll
        for (int mt = 0; mt < 2; ++mt) {
            int r = wm * 32 + mt * 16 + l15;         // A row, m = lane&15
            af[mt].q[0] = *(const u32x4*)&As[r * 32 + hi * 8];
            af[mt].q[1] = *(const u32x4*)&As[r * 32 + 16 + hi * 8];
        }
#pragma unroll
        for (int nt = 0; nt < 4; ++nt) {
            int r = wn * 64 + nt * 16 + l15;         // B column = W row
            bf2[nt].q[0] = *(const u32x4*)&Bs[r * 32 + hi * 8];
            bf2[nt].q[1] = *(const u32x4*)&Bs[r * 32 + 16 + hi * 8];
        }
#pragma unroll
        for (int mt = 0; mt < 2; ++mt)
#pragma unroll
            for (int nt = 0; nt < 4; ++nt)
                c[mt][nt] = wmma_bf16(af[mt].v, bf2[nt].v, c[mt][nt]);
    }

    // epilogue: C layout element (m = j + 8*hi, n = lane&15)
#pragma unroll
    for (int mt = 0; mt < 2; ++mt)
#pragma unroll
        for (int nt = 0; nt < 4; ++nt)
#pragma unroll
            for (int j = 0; j < 8; ++j) {
                int m = mb + wm * 32 + mt * 16 + j + 8 * hi;
                int n = nb + wn * 64 + nt * 16 + l15;
                if (F32OUT)
                    ((float*)O)[(size_t)m * N + n] = c[mt][nt][j];
                else
                    ((__bf16*)O)[(size_t)m * N + n] = (__bf16)c[mt][nt][j];
            }
}

// ---------- kernel 2: RoPE in-place on bf16 [B*S, D], optional out-scale ----------
__global__ void rope_bf16(__bf16* __restrict__ x, float scale) {
    int idx = blockIdx.x * blockDim.x + threadIdx.x;   // one rotation pair
    int row = idx >> 9;             // / (D/2) = /512
    int pr  = idx & 511;
    int s   = row & (PS - 1);       // position within sequence
    int p   = pr & 31;              // pair index within head (d_k/2 = 32)
    int col = ((pr >> 5) << 6) + (p << 1);   // head*64 + 2*p
    // inv_freq = theta^(-2p/64); ln(10000) = 9.2103403719...
    float inv = __expf(-(float)(2 * p) * (9.210340371976184f / (float)PDK));
    float ang = (float)s * inv;
    float sn, cs;
    __sincosf(ang, &sn, &cs);
    size_t base = (size_t)row * PD + col;
    float x1 = (float)x[base];
    float x2 = (float)x[base + 1];
    x[base]     = (__bf16)((x1 * cs - x2 * sn) * scale);
    x[base + 1] = (__bf16)((x1 * sn + x2 * cs) * scale);
}

// ---------- kernel 3: causal flash attention, WMMA bf16 ----------
// 64-key staged blocks, computed as two 32-key halves to bound VGPR pressure.
// grid: (S/128, B*H). block = 256 threads = 8 waves; wave w owns 16 query rows.
__global__ __launch_bounds__(256, 1)
void flash_attn_wmma(const __bf16* __restrict__ Q,
                     const __bf16* __restrict__ Kg,
                     const __bf16* __restrict__ V,
                     __bf16* __restrict__ O) {
    __shared__ __bf16 Ks[64 * 64];        // 64 keys x 64 dims       (8 KB)
    __shared__ __bf16 Vts[64 * 64];       // transposed: dim x key   (8 KB)
    __shared__ __bf16 Ps[8][16 * 32];     // per-wave P tile         (8 KB)

    const int b  = blockIdx.y >> 4;
    const int h  = blockIdx.y & 15;
    const int qb_blk = blockIdx.x * 128;

    const int tid  = threadIdx.x;
    const int lane = tid & 31;
    const int l15  = lane & 15;
    const int hi   = lane >> 4;
    const int w    = tid >> 5;
    const int qb   = qb_blk + w * 16;     // this wave's 16 query rows

    const size_t bh = (size_t)b * PS * PD + (size_t)h * PDK;
    const __bf16* Qb = Q + bh;
    const __bf16* Kb = Kg + bh;
    const __bf16* Vb = V + bh;

    // Q fragments (pre-scaled by 0.125*log2e at RoPE). A layout: m = lane&15.
    Frag qf[2];
#pragma unroll
    for (int dc = 0; dc < 2; ++dc) {
        const __bf16* qrow = Qb + (size_t)(qb + l15) * PD + dc * 32;
        qf[dc].q[0] = *(const u32x4*)(qrow + hi * 8);
        qf[dc].q[1] = *(const u32x4*)(qrow + 16 + hi * 8);
    }

    v8f acc[4] = {};
    float rmax[8], rsum[8];
#pragma unroll
    for (int j = 0; j < 8; ++j) { rmax[j] = -3.0e38f; rsum[j] = 0.0f; }

    const int ktmax = qb_blk + 127;       // block-uniform causal bound
    for (int kt = 0; kt <= ktmax; kt += 64) {
        __syncthreads();                   // prior iteration's reads done
        // stage K tile [key][dim] with b128 loads (512 x 16B)
        for (int i = tid; i < 512; i += 256) {
            int key = i >> 3, qw = i & 7;
            ((u32x4*)Ks)[key * 8 + qw] =
                *(const u32x4*)(Kb + (size_t)(kt + key) * PD + qw * 8);
        }
        // stage V transposed [dim][key]: b128 read, 8x b16 scatter stores
        for (int i = tid; i < 512; i += 256) {
            int key = i >> 3, qw = i & 7;
            Pack8 pw;
            pw.u = *(const u32x4*)(Vb + (size_t)(kt + key) * PD + qw * 8);
#pragma unroll
            for (int d = 0; d < 8; ++d)
                Vts[(qw * 8 + d) * 64 + key] = pw.h[d];
        }
        __syncthreads();

        if (kt > qb + 15) continue;        // wave-uniform: block fully masked

#pragma unroll
        for (int hf = 0; hf < 2; ++hf) {
            const int k0 = kt + hf * 32;
            if (k0 > qb + 15) continue;    // wave-uniform: half fully masked

            // ---- scores: S(16q x 32k), two 16-key C tiles ----
            v8f s0 = {}, s1 = {};
#pragma unroll
            for (int dc = 0; dc < 2; ++dc) {
                Frag kf;
                int r0 = (hf * 32 + l15) * 64 + dc * 32 + hi * 8;
                kf.q[0] = *(const u32x4*)&Ks[r0];
                kf.q[1] = *(const u32x4*)&Ks[r0 + 16];
                s0 = wmma_bf16(qf[dc].v, kf.v, s0);
                int r1 = (hf * 32 + 16 + l15) * 64 + dc * 32 + hi * 8;
                kf.q[0] = *(const u32x4*)&Ks[r1];
                kf.q[1] = *(const u32x4*)&Ks[r1 + 16];
                s1 = wmma_bf16(qf[dc].v, kf.v, s1);
            }

            // ---- causal mask + online softmax (exp2 domain) ----
            __bf16* P = Ps[w];
#pragma unroll
            for (int j = 0; j < 8; ++j) {
                int qg = qb + j + 8 * hi;
                float a  = s0[j];
                float bb = s1[j];
                if (k0 + l15 > qg)      a  = -1.0e30f;
                if (k0 + 16 + l15 > qg) bb = -1.0e30f;
                float mx = redmax16(fmaxf(a, bb));
                float mn   = fmaxf(rmax[j], mx);
                float corr = exp2f(rmax[j] - mn);
                rmax[j] = mn;
                float p0 = exp2f(a - mn);
                float p1 = exp2f(bb - mn);
                float ps = redsum16(p0 + p1);
                rsum[j] = rsum[j] * corr + ps;
                int m = j + 8 * hi;
                P[m * 32 + l15]      = (__bf16)p0;
                P[m * 32 + 16 + l15] = (__bf16)p1;
#pragma unroll
                for (int nt = 0; nt < 4; ++nt)
                    acc[nt][j] *= corr;
            }

            // ---- P fragment (A layout, 16x32) ----
            Frag pf;
            pf.q[0] = *(const u32x4*)&P[l15 * 32 + hi * 8];
            pf.q[1] = *(const u32x4*)&P[l15 * 32 + 16 + hi * 8];

            // ---- PV: out(16x64) += P(16x32) . V(32x64) ----
#pragma unroll
            for (int nt = 0; nt < 4; ++nt) {
                Frag vf;
                int dr = (nt * 16 + l15) * 64 + hf * 32;
                vf.q[0] = *(const u32x4*)&Vts[dr + hi * 8];
                vf.q[1] = *(const u32x4*)&Vts[dr + 16 + hi * 8];
                acc[nt] = wmma_bf16(pf.v, vf.v, acc[nt]);
            }
        }
    }

    // ---- normalize and store (concat-head layout [B*S, D]) ----
#pragma unroll
    for (int j = 0; j < 8; ++j) {
        float inv = 1.0f / rsum[j];
        int q = qb + j + 8 * hi;
        size_t base = ((size_t)b * PS + q) * PD + (size_t)h * PDK;
#pragma unroll
        for (int nt = 0; nt < 4; ++nt)
            O[base + nt * 16 + l15] = (__bf16)(acc[nt][j] * inv);
    }
}

// ---------- host launcher ----------
extern "C" void kernel_launch(void* const* d_in, const int* in_sizes, int n_in,
                              void* d_out, int out_size, void* d_ws, size_t ws_size,
                              hipStream_t stream) {
    const float* X  = (const float*)d_in[0];   // [2,2048,1024]
    const float* Wq = (const float*)d_in[1];   // [1024,1024]
    const float* Wk = (const float*)d_in[2];
    const float* Wv = (const float*)d_in[3];
    const float* Wo = (const float*)d_in[4];
    float* out = (float*)d_out;

    const int NX = PM * PD;        // 4,194,304
    const int NW = PD * PD;        // 1,048,576

    __bf16* p   = (__bf16*)d_ws;
    __bf16* Xb  = p;               p += NX;
    __bf16* Wqb = p;               p += NW;
    __bf16* Wkb = p;               p += NW;
    __bf16* Wvb = p;               p += NW;
    __bf16* Wob = p;               p += NW;
    __bf16* Qb  = p;               p += NX;
    __bf16* Kb  = p;               p += NX;
    __bf16* Vb  = p;               p += NX;
    __bf16* Ab  = p;               p += NX;   // total 48 MB of d_ws

    // 0) fp32 -> bf16 (x4 vectorized)
    cvt_f32_to_bf16x4<<<(NX / 4 + 255) / 256, 256, 0, stream>>>(X,  Xb,  NX / 4);
    cvt_f32_to_bf16x4<<<(NW / 4 + 255) / 256, 256, 0, stream>>>(Wq, Wqb, NW / 4);
    cvt_f32_to_bf16x4<<<(NW / 4 + 255) / 256, 256, 0, stream>>>(Wk, Wkb, NW / 4);
    cvt_f32_to_bf16x4<<<(NW / 4 + 255) / 256, 256, 0, stream>>>(Wv, Wvb, NW / 4);
    cvt_f32_to_bf16x4<<<(NW / 4 + 255) / 256, 256, 0, stream>>>(Wo, Wob, NW / 4);

    // 1) fused QKV projection: grid.z picks {Wq,Wk,Wv} -> {Qb,Kb,Vb}
    dim3 gqkv(PD / GBN, PM / GBM, 3);   // (8, 32, 3)
    gemm_bf16_wmma<false><<<gqkv, 256, 0, stream>>>(
        Xb, Wqb, Wkb, Wvb, Qb, Kb, Vb, PM, PD, PD);

    // 2) RoPE; Q also absorbs (1/sqrt(dk))*log2(e) for exp2-domain softmax
    const int NP = PM * (PD / 2);
    rope_bf16<<<NP / 256, 256, 0, stream>>>(Qb, QSCALE);
    rope_bf16<<<NP / 256, 256, 0, stream>>>(Kb, 1.0f);

    // 3) causal flash attention (64-key staged blocks, 16 WMMA per barrier)
    dim3 gatt(PS / 128, PB * PH);       // (16, 32)
    flash_attn_wmma<<<gatt, 256, 0, stream>>>(Qb, Kb, Vb, Ab);

    // 4) output projection, fp32 store to d_out
    dim3 go(PD / GBN, PM / GBM, 1);
    gemm_bf16_wmma<true><<<go, 256, 0, stream>>>(
        Ab, Wob, Wob, Wob, out, out, out, PM, PD, PD);
}
